// GE2ELoss_75058848465129
// MI455X (gfx1250) — compile-verified, matrix operand order checked by reading, not run on hardware
//
#include <hip/hip_runtime.h>
#include <math.h>

// Problem sizes fixed by setup_inputs(): S=512, U=40, D=1024.
#define S_ 512
#define U_ 40
#define D_ 1024
#define M_ (S_ * U_)   // 20480 rows
#define EPS_ 1e-8f

typedef __attribute__((ext_vector_type(16))) __bf16 v16bf;
typedef __attribute__((ext_vector_type(8)))  __bf16 v8bf;
typedef __attribute__((ext_vector_type(4)))  __bf16 v4bf;
typedef __attribute__((ext_vector_type(8)))  float  v8f;

// ---------------------------------------------------------------------------
// WMMA 16x16x32 bf16 fragment load (wave32 layout per ISA 7.12.2):
// lane L: row = L&15, k-base kb = (L>>4)*8; elements 0..7 -> k=kb..kb+7,
// elements 8..15 -> k=16+kb..16+kb+7.  Two contiguous 16B chunks per frag.
// ---------------------------------------------------------------------------
__device__ inline v16bf load_frag_bf16(const __bf16* __restrict__ p) {
    v8bf c0 = *reinterpret_cast<const v8bf*>(p);
    v8bf c1 = *reinterpret_cast<const v8bf*>(p + 16);
    v16bf r;
#pragma unroll
    for (int i = 0; i < 8; ++i) { r[i] = c0[i]; r[i + 8] = c1[i]; }
    return r;
}

__device__ inline v8f wmma_bf16(v16bf a, v16bf b, v8f c) {
    return __builtin_amdgcn_wmma_f32_16x16x32_bf16(
        false, a, false, b, (short)0, c, false, false);
}

// ---------------------------------------------------------------------------
// Kernel 0: split E (fp32) into bf16 hi + residual lo arrays (A operand).
// One pass, fully coalesced: float4 in, 2x 8-byte bf16x4 out per thread.
// ---------------------------------------------------------------------------
__global__ void __launch_bounds__(256) ge2e_split_a(
    const float* __restrict__ E, __bf16* __restrict__ Ah,
    __bf16* __restrict__ Al) {
    const size_t i = ((size_t)blockIdx.x * 256 + threadIdx.x) * 4;
    const float4 v = *reinterpret_cast<const float4*>(E + i);
    float a[4] = {v.x, v.y, v.z, v.w};
    v4bf h, l;
#pragma unroll
    for (int j = 0; j < 4; ++j) {
        __bf16 hb = (__bf16)a[j];              // RTNE
        h[j] = hb;
        l[j] = (__bf16)(a[j] - (float)hb);     // residual
    }
    *reinterpret_cast<v4bf*>(Ah + i) = h;
    *reinterpret_cast<v4bf*>(Al + i) = l;
}

// ---------------------------------------------------------------------------
// Kernel 1: per-speaker centroid sums Csum[s,:], ||Csum||^2, invC, and the
// bf16 hi/lo split of Csum (B operand of the big GEMM).
// One block per speaker; thread t owns columns 4t..4t+3.
// ---------------------------------------------------------------------------
__global__ void __launch_bounds__(256) ge2e_centroid(
    const float* __restrict__ E, float* __restrict__ Csum,
    float* __restrict__ cnorm2, float* __restrict__ invC,
    __bf16* __restrict__ Bh, __bf16* __restrict__ Bl) {
    __shared__ float red[256];
    const int s = blockIdx.x;
    const int t = threadIdx.x;
    const int d0 = t * 4;
    float4 acc = {0.f, 0.f, 0.f, 0.f};
    for (int u = 0; u < U_; ++u) {
        const float4 v = *reinterpret_cast<const float4*>(
            E + (size_t)(s * U_ + u) * D_ + d0);
        acc.x += v.x; acc.y += v.y; acc.z += v.z; acc.w += v.w;
    }
    *reinterpret_cast<float4*>(Csum + (size_t)s * D_ + d0) = acc;

    float av[4] = {acc.x, acc.y, acc.z, acc.w};
    v4bf h, l;
#pragma unroll
    for (int i = 0; i < 4; ++i) {
        __bf16 hb = (__bf16)av[i];
        h[i] = hb;
        l[i] = (__bf16)(av[i] - (float)hb);
    }
    *reinterpret_cast<v4bf*>(Bh + (size_t)s * D_ + d0) = h;
    *reinterpret_cast<v4bf*>(Bl + (size_t)s * D_ + d0) = l;

    red[t] = av[0]*av[0] + av[1]*av[1] + av[2]*av[2] + av[3]*av[3];
    __syncthreads();
    for (int off = 128; off > 0; off >>= 1) {
        if (t < off) red[t] += red[t + off];
        __syncthreads();
    }
    if (t == 0) {
        float n2 = red[0];
        cnorm2[s] = n2;
        // cin = (Csum/U)/max(||Csum||/U, eps)  =>  invC = 1/max(||Csum||, U*eps)
        invC[s] = 1.0f / fmaxf(sqrtf(n2), (float)U_ * EPS_);
    }
}

// ---------------------------------------------------------------------------
// Kernel 2: per-utterance-row stats: ||e||, dot(e, Csum_s), and sim_own via
// the exclusive-centroid identities. One wave32 per row (8 rows per block).
// ---------------------------------------------------------------------------
__global__ void __launch_bounds__(256) ge2e_rowstats(
    const float* __restrict__ E, const float* __restrict__ Csum,
    const float* __restrict__ cnorm2,
    float* __restrict__ invE, float* __restrict__ simOwn) {
    const int row  = blockIdx.x * 8 + (threadIdx.x >> 5);
    const int lane = threadIdx.x & 31;
    const int s    = row / U_;
    const float* e = E + (size_t)row * D_;
    const float* c = Csum + (size_t)s * D_;
    float n2 = 0.f, dc = 0.f;
#pragma unroll 4
    for (int j = 0; j < D_ / 32; ++j) {
        float x = e[lane + 32 * j];
        float y = c[lane + 32 * j];
        n2 += x * x;
        dc += x * y;
    }
    for (int m = 16; m; m >>= 1) {
        n2 += __shfl_xor(n2, m, 32);
        dc += __shfl_xor(dc, m, 32);
    }
    if (lane == 0) {
        float ie = 1.0f / fmaxf(sqrtf(n2), EPS_);
        invE[row] = ie;
        float cex2 = fmaxf(cnorm2[s] - 2.f * dc + n2, 0.f);  // ||Csum - e||^2
        float icex = 1.0f / fmaxf(sqrtf(cex2), (float)(U_ - 1) * EPS_);
        simOwn[row] = (dc - n2) * ie * icex;
    }
}

// ---------------------------------------------------------------------------
// Kernel 3: raw-dot GEMM  Cmat[M,S] = E[M,D] * Csum[S,D]^T  via split-fp32
// bf16 WMMA (hi*hi + lo*hi + hi*lo).  Pure load+WMMA inner loop.
// Each wave: 32M x 64N strip (2 m-tiles x 4 n-tiles, 8 accumulators);
// B fragments reused across both m-tiles.  Block = 4 waves -> 128M x 64N.
// grid = (M/128, S/64).  Per k-slab: 24 b128 loads vs 24 WMMAs.
// ---------------------------------------------------------------------------
__global__ void __launch_bounds__(128) ge2e_gemm_wmma(
    const __bf16* __restrict__ Ah, const __bf16* __restrict__ Al,
    const __bf16* __restrict__ Bh, const __bf16* __restrict__ Bl,
    float* __restrict__ C) {
    const int lane = threadIdx.x & 31;
    const int wave = threadIdx.x >> 5;
    const int m0 = (blockIdx.x * 8 + wave * 2) * 16;   // m-tiles m0, m0+16
    const int n0 = blockIdx.y * 64;
    const int mr = lane & 15;
    const int kb = (lane >> 4) * 8;

    v8f acc[2][4] = {{v8f{}, v8f{}, v8f{}, v8f{}},
                     {v8f{}, v8f{}, v8f{}, v8f{}}};

    const size_t aoff0 = (size_t)(m0 + mr) * D_ + kb;
    const size_t aoff1 = aoff0 + (size_t)16 * D_;
    const size_t boff  = (size_t)(n0 + mr) * D_ + kb;

    for (int k0 = 0; k0 < D_; k0 += 32) {
        const v16bf ah0 = load_frag_bf16(Ah + aoff0 + k0);
        const v16bf al0 = load_frag_bf16(Al + aoff0 + k0);
        const v16bf ah1 = load_frag_bf16(Ah + aoff1 + k0);
        const v16bf al1 = load_frag_bf16(Al + aoff1 + k0);
#pragma unroll
        for (int nt = 0; nt < 4; ++nt) {
            const size_t o = boff + (size_t)nt * 16 * D_ + k0;
            const v16bf bh = load_frag_bf16(Bh + o);
            const v16bf bl = load_frag_bf16(Bl + o);
            acc[0][nt] = wmma_bf16(ah0, bh, acc[0][nt]);
            acc[0][nt] = wmma_bf16(al0, bh, acc[0][nt]);
            acc[0][nt] = wmma_bf16(ah0, bl, acc[0][nt]);
            acc[1][nt] = wmma_bf16(ah1, bh, acc[1][nt]);
            acc[1][nt] = wmma_bf16(al1, bh, acc[1][nt]);
            acc[1][nt] = wmma_bf16(ah1, bl, acc[1][nt]);
        }
    }

    // D layout: VGPR r, lanes 0-15 -> M=r, lanes 16-31 -> M=8+r; N = lane&15.
    const int nc = lane & 15;
#pragma unroll
    for (int mt = 0; mt < 2; ++mt) {
        const int mbase = m0 + mt * 16 + (lane >> 4) * 8;
#pragma unroll
        for (int nt = 0; nt < 4; ++nt) {
#pragma unroll
            for (int r = 0; r < 8; ++r) {
                C[(size_t)(mbase + r) * S_ + (n0 + nt * 16 + nc)] =
                    acc[mt][nt][r];
            }
        }
    }
}

// ---------------------------------------------------------------------------
// Kernel 4: per-row scale -> logits -> logsumexp -> (logz - target logit).
// One wave32 per row, 8 rows/block; fixed-order reductions (deterministic).
// ---------------------------------------------------------------------------
__global__ void __launch_bounds__(256) ge2e_rowce(
    const float* __restrict__ Cmat, const float* __restrict__ invE,
    const float* __restrict__ invC, const float* __restrict__ simOwn,
    const float* __restrict__ wp, const float* __restrict__ bp,
    float* __restrict__ partials) {
    __shared__ float red[8];
    const int row  = blockIdx.x * 8 + (threadIdx.x >> 5);
    const int lane = threadIdx.x & 31;
    const float w = *wp, b = *bp;
    const int s = row / U_;
    const float ie = invE[row];
    const float so = simOwn[row];
    const float* crow = Cmat + (size_t)row * S_;

    float l[S_ / 32];
    float mx = -3.0e38f;
#pragma unroll
    for (int j = 0; j < S_ / 32; ++j) {
        int k = lane + 32 * j;
        float sim = (k == s) ? so : crow[k] * ie * invC[k];
        float lg = w * sim + b;
        l[j] = lg;
        mx = fmaxf(mx, lg);
    }
    for (int m = 16; m; m >>= 1) mx = fmaxf(mx, __shfl_xor(mx, m, 32));
    float se = 0.f;
#pragma unroll
    for (int j = 0; j < S_ / 32; ++j) se += __expf(l[j] - mx);
    for (int m = 16; m; m >>= 1) se += __shfl_xor(se, m, 32);
    if (lane == 0) {
        float logz = mx + __logf(se);
        red[threadIdx.x >> 5] = logz - (w * so + b);
    }
    __syncthreads();
    if (threadIdx.x == 0) {
        float t = 0.f;
#pragma unroll
        for (int i = 0; i < 8; ++i) t += red[i];
        partials[blockIdx.x] = t;
    }
}

// ---------------------------------------------------------------------------
// Kernel 5: deterministic final reduction -> mean -> d_out[0].
// ---------------------------------------------------------------------------
__global__ void __launch_bounds__(256) ge2e_final(
    const float* __restrict__ partials, int n, float* __restrict__ out) {
    __shared__ float red[256];
    float a = 0.f;
    for (int i = threadIdx.x; i < n; i += 256) a += partials[i];
    red[threadIdx.x] = a;
    __syncthreads();
    for (int off = 128; off > 0; off >>= 1) {
        if (threadIdx.x < off) red[threadIdx.x] += red[threadIdx.x + off];
        __syncthreads();
    }
    if (threadIdx.x == 0) out[0] = red[0] / (float)M_;
}

// ---------------------------------------------------------------------------
extern "C" void kernel_launch(void* const* d_in, const int* in_sizes, int n_in,
                              void* d_out, int out_size, void* d_ws, size_t ws_size,
                              hipStream_t stream) {
    (void)in_sizes; (void)n_in; (void)out_size; (void)ws_size;
    const float* E  = (const float*)d_in[0];
    const float* wp = (const float*)d_in[1];
    const float* bp = (const float*)d_in[2];
    // d_in[3], d_in[4] are S and U (device ints); sizes are compile-time here.

    // Workspace carve-up (~130 MB total).
    float* Csum     = (float*)d_ws;                      // S*D
    float* cnorm2   = Csum + (size_t)S_ * D_;            // S
    float* invC     = cnorm2 + S_;                       // S
    float* invE     = invC + S_;                         // M
    float* simOwn   = invE + M_;                         // M
    float* partials = simOwn + M_;                       // M/8
    float* Cmat     = partials + (M_ / 8);               // M*S  (40 MB)
    __bf16* Bh      = (__bf16*)(Cmat + (size_t)M_ * S_); // S*D
    __bf16* Bl      = Bh + (size_t)S_ * D_;              // S*D
    __bf16* Ah      = Bl + (size_t)S_ * D_;              // M*D  (42 MB)
    __bf16* Al      = Ah + (size_t)M_ * D_;              // M*D  (42 MB)

    ge2e_split_a<<<(int)(((size_t)M_ * D_) / 1024), 256, 0, stream>>>(E, Ah, Al);
    ge2e_centroid<<<S_, 256, 0, stream>>>(E, Csum, cnorm2, invC, Bh, Bl);
    ge2e_rowstats<<<M_ / 8, 256, 0, stream>>>(E, Csum, cnorm2, invE, simOwn);
    dim3 gg(M_ / 128, S_ / 64);
    ge2e_gemm_wmma<<<gg, 128, 0, stream>>>(Ah, Al, Bh, Bl, Cmat);
    ge2e_rowce<<<M_ / 8, 256, 0, stream>>>(Cmat, invE, invC, simOwn, wp, bp,
                                           partials);
    ge2e_final<<<1, 256, 0, stream>>>(partials, M_ / 8, (float*)d_out);
}
